// FullAttention_28879360098586
// MI455X (gfx1250) — matrix operand order repaired and checked by reading, not compiled, not run on hardware
//
#include <hip/hip_runtime.h>
#include <math.h>

// MI455X / gfx1250 flash-attention with multiplicative adaptive weights.
// bf16 WMMA (v_wmma_f32_16x16x32_bf16) for both GEMMs, fp32 softmax math.
// Workgroup = 256 threads (8 wave32), each wave owns 16 query rows.
// Grid = B*H*(V/128) = 512 blocks. S processed in 64-key chunks.

typedef __attribute__((ext_vector_type(16))) __bf16 v16bf;
typedef __attribute__((ext_vector_type(2)))  __bf16 v2bf;
typedef __attribute__((ext_vector_type(8)))  float  v8f;

union FragBF { v16bf v; unsigned int u[8]; };
union FragF  { v8f   v; float        f[8]; };

__device__ __forceinline__ unsigned short f2bf(float x) {
  __bf16 h = (__bf16)x;                      // native v_cvt, RNE
  return __builtin_bit_cast(unsigned short, h);
}
__device__ __forceinline__ unsigned int pack2bf(float a, float b) {
  v2bf p; p.x = (__bf16)a; p.y = (__bf16)b;  // v_cvt_pk_bf16_f32
  return __builtin_bit_cast(unsigned int, p);
}

constexpr int   B_ = 8, V_ = 1024, S_ = 1024, H_ = 8, E_ = 64, D_ = 64;
constexpr int   SC = 64;            // keys per chunk
constexpr float NEG_INF = -1e30f;
constexpr float SCALE   = 0.125f;   // 1/sqrt(E)

__global__ __launch_bounds__(256)
void fullattn_adaptive_kernel(const float* __restrict__ Q,
                              const float* __restrict__ K,
                              const float* __restrict__ Vv,
                              const unsigned char* __restrict__ Mk,  // bool mask [B,1,V,S]
                              const float* __restrict__ W,           // [B,H,V,S]
                              float* __restrict__ O)                 // [B,V,H,D]
{
  __shared__ unsigned int   Kl[SC * 32];      // 64 keys x 64 e, bf16 pairs (8 KB)
  __shared__ unsigned short Vt[64 * SC];      // transposed values [d][k] bf16 (8 KB)
  __shared__ unsigned short Pl[8][16 * SC];   // per-wave P tile [m][k] bf16 (16 KB)

  const int tid  = threadIdx.x;
  const int wave = tid >> 5;
  const int lane = tid & 31;
  const int half = lane >> 4;   // 0: lanes 0-15, 1: lanes 16-31
  const int ln   = lane & 15;

  const int blk = blockIdx.x;
  const int vt  = blk & 7;          // V tile (V/128 = 8)
  const int h   = (blk >> 3) & 7;   // head  (H = 8)
  const int b   = blk >> 6;         // batch
  const int v0  = vt * 128 + wave * 16;   // first query row of this wave

  // ---- Q fragments: two 16x32 bf16 A-matrices, resident all loop ----
  // A layout (16-bit 16x32): lane holds row M=ln;
  // VGPR j: K = (j<4 ? 2j : 8+2j) + 8*half  (+32 for second fragment)
  FragBF qf0, qf1;
  {
    const int qrow = v0 + ln;
    const float* qp = Q + (((size_t)(b * V_ + qrow)) * H_ + h) * E_;
#pragma unroll
    for (int j = 0; j < 8; ++j) {
      const int e = ((j < 4) ? 2 * j : 8 + 2 * j) + half * 8;
      qf0.u[j] = pack2bf(qp[e],      qp[e + 1]);
      qf1.u[j] = pack2bf(qp[e + 32], qp[e + 33]);
    }
  }

  // per-row online-softmax state; C-layout row mapping: i -> M = i + 8*half
  float m[8], l[8];
  FragF acc[4];
#pragma unroll
  for (int i = 0; i < 8; ++i) { m[i] = -3.0e38f; l[i] = 0.0f; }
#pragma unroll
  for (int t = 0; t < 4; ++t)
#pragma unroll
    for (int i = 0; i < 8; ++i) acc[t].f[i] = 0.0f;

  const size_t wRow = ((size_t)(b * H_ + h)) * V_;   // weight row base
  const size_t mRow = (size_t)b * V_;                // mask row base
  const v8f cz = {0.f, 0.f, 0.f, 0.f, 0.f, 0.f, 0.f, 0.f};

  for (int s0 = 0; s0 < S_; s0 += SC) {
    // ---- cooperative staging: K chunk (bf16 row-major), V chunk (bf16 [d][k]) ----
    {
      const int srow = tid >> 2;            // 0..63 (4 threads per key row)
      const int c0   = (tid & 3) * 16;      // 16 contiguous elements per thread
      const float* kp = K + (((size_t)(b * S_ + s0 + srow)) * H_ + h) * E_ + c0;
      const float* vp = Vv + (((size_t)(b * S_ + s0 + srow)) * H_ + h) * D_ + c0;
#pragma unroll
      for (int q = 0; q < 8; ++q)
        Kl[srow * 32 + (c0 >> 1) + q] = pack2bf(kp[2 * q], kp[2 * q + 1]);
#pragma unroll
      for (int ii = 0; ii < 16; ++ii)
        Vt[(c0 + ii) * SC + srow] = f2bf(vp[ii]);
    }
    __syncthreads();

    // ---- scores: 16x64 tile as four 16x16 WMMA C-tiles, E=64 contraction ----
    float lg[4][8];
#pragma unroll
    for (int nt = 0; nt < 4; ++nt) {
      const int n0  = nt * 16;
      const int col = n0 + ln;             // key index in chunk (B-matrix N = ln)
      FragBF kf0, kf1;
#pragma unroll
      for (int j = 0; j < 8; ++j) {
        // B layout (16-bit 32x16): VGPR j holds K = (j<4 ? 2j : 8+2j) + 8*half, +1
        const int e = ((j < 4) ? 2 * j : 8 + 2 * j) + half * 8;
        kf0.u[j] = Kl[(col * 64 + e) >> 1];
        kf1.u[j] = Kl[(col * 64 + e + 32) >> 1];
      }
      FragF c;
      c.v = __builtin_amdgcn_wmma_f32_16x16x32_bf16(false, qf0.v, false, kf0.v,
                                                    (short)0, cz, false, false);
      c.v = __builtin_amdgcn_wmma_f32_16x16x32_bf16(false, qf1.v, false, kf1.v,
                                                    (short)0, c.v, false, false);
      // mask + adaptive weight + scale -> logits (exact reference order)
#pragma unroll
      for (int i = 0; i < 8; ++i) {
        const int row  = v0 + i + 8 * half;
        const int scol = s0 + n0 + ln;
        const float qk  = c.f[i];
        const float wgt = W[(wRow + row) * S_ + scol];
        const unsigned char mk = Mk[(mRow + row) * S_ + scol];
        const float sc = mk ? NEG_INF : qk;
        lg[nt][i] = SCALE * (sc * wgt);
      }
    }

    // ---- online softmax (per row: 16-lane reduction within each half) ----
#pragma unroll
    for (int i = 0; i < 8; ++i) {
      float mx = fmaxf(fmaxf(lg[0][i], lg[1][i]), fmaxf(lg[2][i], lg[3][i]));
#pragma unroll
      for (int off = 1; off < 16; off <<= 1)
        mx = fmaxf(mx, __shfl_xor(mx, off, 32));
      const float nm   = fmaxf(m[i], mx);
      const float corr = __expf(m[i] - nm);
      m[i] = nm;
      float e[4], ps = 0.0f;
#pragma unroll
      for (int nt = 0; nt < 4; ++nt) { e[nt] = __expf(lg[nt][i] - nm); ps += e[nt]; }
#pragma unroll
      for (int off = 1; off < 16; off <<= 1)
        ps += __shfl_xor(ps, off, 32);
      l[i] = l[i] * corr + ps;
#pragma unroll
      for (int t = 0; t < 4; ++t) acc[t].f[i] *= corr;
      // stash P (bf16) for layout transpose through LDS
#pragma unroll
      for (int nt = 0; nt < 4; ++nt)
        Pl[wave][(i + 8 * half) * SC + nt * 16 + ln] = f2bf(e[nt]);
    }
    __syncthreads();

    // ---- P·V: P as two 16x32 A-fragments, V as 32x16 B-fragments ----
    FragBF pf0, pf1;
#pragma unroll
    for (int j = 0; j < 8; ++j) {
      const int k = ((j < 4) ? 2 * j : 8 + 2 * j) + half * 8;
      pf0.u[j] = *(const unsigned int*)&Pl[wave][ln * SC + k];
      pf1.u[j] = *(const unsigned int*)&Pl[wave][ln * SC + k + 32];
    }
#pragma unroll
    for (int t = 0; t < 4; ++t) {
      const int d = t * 16 + ln;
      FragBF vf0, vf1;
#pragma unroll
      for (int j = 0; j < 8; ++j) {
        const int k = ((j < 4) ? 2 * j : 8 + 2 * j) + half * 8;
        vf0.u[j] = *(const unsigned int*)&Vt[d * SC + k];
        vf1.u[j] = *(const unsigned int*)&Vt[d * SC + k + 32];
      }
      acc[t].v = __builtin_amdgcn_wmma_f32_16x16x32_bf16(false, pf0.v, false, vf0.v,
                                                         (short)0, acc[t].v,
                                                         false, false);
      acc[t].v = __builtin_amdgcn_wmma_f32_16x16x32_bf16(false, pf1.v, false, vf1.v,
                                                         (short)0, acc[t].v,
                                                         false, false);
    }
    __syncthreads();   // protect K/V/P LDS before next chunk overwrites
  }

  // ---- epilogue: normalize and store [B,V,H,D] ----
#pragma unroll
  for (int i = 0; i < 8; ++i) {
    const float inv = 1.0f / l[i];
    const int row = v0 + i + 8 * half;
    float* op = O + (((size_t)(b * V_ + row)) * H_ + h) * D_;
#pragma unroll
    for (int t = 0; t < 4; ++t)
      op[t * 16 + ln] = acc[t].f[i] * inv;
  }
}

extern "C" void kernel_launch(void* const* d_in, const int* in_sizes, int n_in,
                              void* d_out, int out_size, void* d_ws, size_t ws_size,
                              hipStream_t stream) {
  (void)in_sizes; (void)n_in; (void)d_ws; (void)ws_size; (void)out_size;
  const float* Q          = (const float*)d_in[0];
  const float* K          = (const float*)d_in[1];
  const float* Vv         = (const float*)d_in[2];
  const unsigned char* Mk = (const unsigned char*)d_in[3];  // jnp bool -> 1 byte
  const float* W          = (const float*)d_in[4];
  float* O                = (float*)d_out;

  dim3 grid(B_ * H_ * (V_ / 128));  // 512
  dim3 block(256);                  // 8 wave32
  hipLaunchKernelGGL(fullattn_adaptive_kernel, grid, block, 0, stream,
                     Q, K, Vv, Mk, W, O);
}
